// MultiHeadAttention_35983236006264
// MI455X (gfx1250) — matrix operand order repaired and checked by reading, not compiled, not run on hardware
//
#include <hip/hip_runtime.h>

// MI455X / gfx1250, wave32. BF16 WMMA (16x16x32) flash attention with
// async global->LDS double-buffered K/V staging.

typedef __attribute__((ext_vector_type(16))) __bf16 v16bf;
typedef __attribute__((ext_vector_type(8)))  __bf16 v8bf;
typedef __attribute__((ext_vector_type(8)))  float  v8f;
typedef __attribute__((ext_vector_type(4)))  float  v4f;
typedef __attribute__((ext_vector_type(4)))  int    v4i;

#define TLEN 2048
#define EDIM 1024
#define HNUM 16
#define DDIM 64
#define BHN  64   // B*H

// ---- CDNA5 async global->LDS (guarded; falls back to load+ds_store) -------
#if defined(__AMDGCN__) && __has_builtin(__builtin_amdgcn_global_load_async_to_lds_b128)
#define HAVE_ASYNC_LDS 1
#else
#define HAVE_ASYNC_LDS 0
#endif

#if HAVE_ASYNC_LDS
typedef __attribute__((address_space(1))) v4i g_v4i;   // global int4
typedef __attribute__((address_space(3))) v4i l_v4i;   // LDS int4
static __device__ __forceinline__ void copy16_async(const __bf16* g, __bf16* l) {
  // (src global int4*, dst lds int4*, imm offset, imm cpol)
  __builtin_amdgcn_global_load_async_to_lds_b128((g_v4i*)g, (l_v4i*)l, 0, 0);
}
#if __has_builtin(__builtin_amdgcn_s_wait_asynccnt)
#define WAIT_ASYNC(n) __builtin_amdgcn_s_wait_asynccnt(n)
#else
#define WAIT_ASYNC(n) asm volatile("s_wait_asynccnt %0" ::"i"(n) : "memory")
#endif
#else
static __device__ __forceinline__ void copy16_async(const __bf16* g, __bf16* l) {
  *(v8bf*)l = *(const v8bf*)g;
}
#define WAIT_ASYNC(n) ((void)0)
#endif

static __device__ __forceinline__ v8f wmma_bf16(v16bf a, v16bf b, v8f c) {
  // (neg_a, A, neg_b, B, c_mod, C, reuse_a, reuse_b)
  return __builtin_amdgcn_wmma_f32_16x16x32_bf16(false, a, false, b, (short)0, c,
                                                 false, false);
}

static __device__ __forceinline__ v16bf cat8(v8bf lo, v8bf hi) {
  return __builtin_shufflevector(lo, hi, 0, 1, 2, 3, 4, 5, 6, 7,
                                 8, 9, 10, 11, 12, 13, 14, 15);
}

static __device__ __forceinline__ float fast_exp2(float x) {
#if defined(__AMDGCN__) && __has_builtin(__builtin_amdgcn_exp2f)
  return __builtin_amdgcn_exp2f(x);
#else
  return exp2f(x);
#endif
}

// ---------------------------------------------------------------------------
// Kernel 1: per-head QKV projection, fp32 in -> bf16 out.
//   q,k : [BH, T, D] row-major bf16
//   v   : stored TRANSPOSED [BH, D, T] bf16
// Block: 128 threads (4 waves); tile 64(T) x 64(D); K-loop over E in 32-chunks.
// ---------------------------------------------------------------------------
__global__ __launch_bounds__(128)
void mha_proj_kernel(const float* __restrict__ x,
                     const float* __restrict__ Wq,
                     const float* __restrict__ Wk,
                     const float* __restrict__ Wv,
                     __bf16* __restrict__ qws,
                     __bf16* __restrict__ kws,
                     __bf16* __restrict__ vtws) {
  __shared__ alignas(16) __bf16 Wt[64][40];  // [d][e-chunk], padded stride

  const int t0 = blockIdx.x * 64;
  const int bh = blockIdx.y;
  const int p  = blockIdx.z;          // 0=q, 1=k, 2=v
  const int b  = bh >> 4;
  const int h  = bh & 15;
  const int tid  = threadIdx.x;
  const int lane = tid & 31;
  const int w    = tid >> 5;
  const int ln16 = lane & 15;
  const int hl   = lane >> 4;
  const int kbase = hl * 8;
  const int khalf = hl * 16;

  const float* W  = (p == 0) ? Wq : (p == 1) ? Wk : Wv;
  const float* Wh = W + (size_t)h * EDIM * DDIM;
  const float* xrow =
      x + ((size_t)b * TLEN + (t0 + w * 16 + ln16)) * EDIM;

  v8f acc[4];
#pragma unroll
  for (int n = 0; n < 4; ++n) acc[n] = (v8f){};

  for (int e0 = 0; e0 < EDIM; e0 += 32) {
    // Stage W tile transposed: Wt[d][e] = Wh[e0+e][d]
#pragma unroll
    for (int i = 0; i < 16; ++i) {
      int linear = tid + 128 * i;     // e*64 + d  -> coalesced global reads
      int d = linear & 63;
      int e = linear >> 6;
      Wt[d][e] = (__bf16)Wh[(size_t)(e0 + e) * DDIM + d];
    }
    __syncthreads();

    // A fragment of x (direct from global, f32 -> bf16)
    v4f f0 = *(const v4f*)(xrow + e0 + kbase);
    v4f f1 = *(const v4f*)(xrow + e0 + kbase + 4);
    v4f f2 = *(const v4f*)(xrow + e0 + 16 + kbase);
    v4f f3 = *(const v4f*)(xrow + e0 + 16 + kbase + 4);
    v16bf a;
#pragma unroll
    for (int q = 0; q < 4; ++q) {
      a[q]      = (__bf16)f0[q];
      a[q + 4]  = (__bf16)f1[q];
      a[q + 8]  = (__bf16)f2[q];
      a[q + 12] = (__bf16)f3[q];
    }

#pragma unroll
    for (int n = 0; n < 4; ++n) {
      v8bf b0 = *(const v8bf*)&Wt[n * 16 + ln16][kbase];
      v8bf b1 = *(const v8bf*)&Wt[n * 16 + ln16][16 + kbase];
      acc[n] = wmma_bf16(a, cat8(b0, b1), acc[n]);
    }
    __syncthreads();
  }

  if (p < 2) {
    __bf16* outp = (p == 0) ? qws : kws;
#pragma unroll
    for (int n = 0; n < 4; ++n) {
      int d = n * 16 + ln16;
#pragma unroll
      for (int r = 0; r < 8; ++r) {
        int row = t0 + w * 16 + r + 8 * hl;
        outp[((size_t)bh * TLEN + row) * DDIM + d] = (__bf16)acc[n][r];
      }
    }
  } else {
    // v transposed: per lane the 8 C-regs are 8 consecutive t -> packed b128
#pragma unroll
    for (int n = 0; n < 4; ++n) {
      int d  = n * 16 + ln16;
      int tb = t0 + w * 16 + hl * 8;
      v8bf pk;
#pragma unroll
      for (int r = 0; r < 8; ++r) pk[r] = (__bf16)acc[n][r];
      *(v8bf*)&vtws[((size_t)bh * DDIM + d) * TLEN + tb] = pk;
    }
  }
}

// ---------------------------------------------------------------------------
// Stage one 64x64 K tile and one 64x64 V^T tile into LDS (padded stride 72).
// 128 threads * 8 b128 granules each. Async on CDNA5 (ASYNCcnt-tracked).
// ---------------------------------------------------------------------------
static __device__ __forceinline__ void stage_kv(const __bf16* __restrict__ kg,
                                                const __bf16* __restrict__ vg,
                                                int k0, __bf16* Kt, __bf16* Vt,
                                                int tid) {
#pragma unroll
  for (int it = 0; it < 4; ++it) {
    int t   = tid + 128 * it;        // 0..511
    int row = t >> 3;                // 0..63  (key)
    int cc  = (t & 7) * 8;           // d offset
    copy16_async(kg + (size_t)(k0 + row) * DDIM + cc, Kt + row * 72 + cc);
  }
#pragma unroll
  for (int it = 0; it < 4; ++it) {
    int t   = tid + 128 * it;
    int row = t >> 3;                // 0..63  (d)
    int cc  = (t & 7) * 8;           // key offset
    copy16_async(vg + (size_t)row * TLEN + k0 + cc, Vt + row * 72 + cc);
  }
}

// ---------------------------------------------------------------------------
// Kernel 2: causal flash attention over bf16 q/k/vT, fp32 accumulate.
// Block: 128 threads (4 waves). Q-tile = 64 rows; wave w owns rows w*16..+15.
// K/V tiles double-buffered in LDS via async copies; softmax in exp2 domain.
// ---------------------------------------------------------------------------
__global__ __launch_bounds__(128)
void mha_attn_kernel(const __bf16* __restrict__ qws,
                     const __bf16* __restrict__ kws,
                     const __bf16* __restrict__ vtws,
                     float* __restrict__ out) {
  __shared__ alignas(16) __bf16 Ktile[2][64][72];
  __shared__ alignas(16) __bf16 Vtile[2][64][72];
  __shared__ alignas(16) __bf16 Pb[4][16][72];

  const int i  = blockIdx.x;          // q tile
  const int bh = blockIdx.y;
  const int b  = bh >> 4;
  const int h  = bh & 15;
  const int tid  = threadIdx.x;
  const int lane = tid & 31;
  const int w    = tid >> 5;
  const int ln16 = lane & 15;
  const int hl   = lane >> 4;
  const int kbase = hl * 8;
  const int khalf = hl * 16;

  const __bf16* kg = kws + (size_t)bh * TLEN * DDIM;
  const __bf16* vg = vtws + (size_t)bh * DDIM * TLEN;

  // Persistent Q A-fragments (two 16x32 K-chunks covering D=64)
  const __bf16* qrow =
      qws + ((size_t)bh * TLEN + (i * 64 + w * 16 + ln16)) * DDIM;
  v16bf qa[2];
#pragma unroll
  for (int c = 0; c < 2; ++c) {
    v8bf q0 = *(const v8bf*)(qrow + 32 * c + kbase);
    v8bf q1 = *(const v8bf*)(qrow + 32 * c + 16 + kbase);
    qa[c] = cat8(q0, q1);
  }

  // log2(e)/sqrt(D): run the online softmax in the exp2 domain
  const float SC = 0.125f * 1.44269504088896341f;

  float m[8], l[8];
  v8f acc[4];
#pragma unroll
  for (int r = 0; r < 8; ++r) { m[r] = -1e30f; l[r] = 0.0f; }
#pragma unroll
  for (int n = 0; n < 4; ++n) acc[n] = (v8f){};

  // Prefetch tile 0
  stage_kv(kg, vg, 0, &Ktile[0][0][0], &Vtile[0][0][0], tid);

  for (int j = 0; j <= i; ++j) {
    const int cur = j & 1;
    if (j < i) {
      stage_kv(kg, vg, (j + 1) * 64, &Ktile[cur ^ 1][0][0],
               &Vtile[cur ^ 1][0][0], tid);
      WAIT_ASYNC(8);   // previous tile landed; next still in flight
    } else {
      WAIT_ASYNC(0);
    }
    __syncthreads();

    // ---- S = (Q * K^T) * SC : K tile row-major [key][d] IS B^T already
    v8f s[4];
#pragma unroll
    for (int n = 0; n < 4; ++n) {
      const __bf16* krow = &Ktile[cur][n * 16 + ln16][0];
      v8f z = (v8f){};
#pragma unroll
      for (int c = 0; c < 2; ++c) {
        v8bf b0 = *(const v8bf*)(krow + 32 * c + kbase);
        v8bf b1 = *(const v8bf*)(krow + 32 * c + 16 + kbase);
        z = wmma_bf16(qa[c], cat8(b0, b1), z);
      }
      s[n] = z;
    }
#pragma unroll
    for (int n = 0; n < 4; ++n)
#pragma unroll
      for (int r = 0; r < 8; ++r) s[n][r] *= SC;

    // ---- causal mask on diagonal tile
    if (j == i) {
#pragma unroll
      for (int n = 0; n < 4; ++n) {
        int col = j * 64 + n * 16 + ln16;
#pragma unroll
        for (int r = 0; r < 8; ++r) {
          int row = i * 64 + w * 16 + r + 8 * hl;
          if (col > row) s[n][r] = -1e30f;
        }
      }
    }

    // ---- online softmax (row stats live in each half-wave's 16 lanes)
#pragma unroll
    for (int r = 0; r < 8; ++r) {
      float loc = fmaxf(fmaxf(s[0][r], s[1][r]), fmaxf(s[2][r], s[3][r]));
#pragma unroll
      for (int off = 1; off < 16; off <<= 1)
        loc = fmaxf(loc, __shfl_xor(loc, off, 16));
      float mn = fmaxf(m[r], loc);
      float alpha = fast_exp2(m[r] - mn);
      m[r] = mn;
      float rs = 0.0f;
#pragma unroll
      for (int n = 0; n < 4; ++n) {
        float pv = fast_exp2(s[n][r] - mn);
        s[n][r] = pv;
        rs += pv;
      }
#pragma unroll
      for (int off = 1; off < 16; off <<= 1)
        rs += __shfl_xor(rs, off, 16);
      l[r] = l[r] * alpha + rs;
#pragma unroll
      for (int n = 0; n < 4; ++n) acc[n][r] *= alpha;
    }

    // ---- P: C-layout regs -> per-wave LDS (bf16) -> A-fragments
#pragma unroll
    for (int n = 0; n < 4; ++n)
#pragma unroll
      for (int r = 0; r < 8; ++r)
        Pb[w][r + 8 * hl][n * 16 + ln16] = (__bf16)s[n][r];

    // ---- O += P * V  (B-fragments from LDS-staged V^T tile)
#pragma unroll
    for (int c = 0; c < 2; ++c) {
      v8bf p0 = *(const v8bf*)&Pb[w][ln16][32 * c + kbase];
      v8bf p1 = *(const v8bf*)&Pb[w][ln16][32 * c + 16 + kbase];
      v16bf pa = cat8(p0, p1);
#pragma unroll
      for (int n = 0; n < 4; ++n) {
        const __bf16* vrow = &Vtile[cur][n * 16 + ln16][0];
        v8bf b0 = *(const v8bf*)(vrow + 32 * c + kbase);
        v8bf b1 = *(const v8bf*)(vrow + 32 * c + 16 + kbase);
        acc[n] = wmma_bf16(pa, cat8(b0, b1), acc[n]);
      }
    }
    __syncthreads();  // protect buffer reuse before next stage_kv
  }

  // ---- epilogue: out[b][t][h*64+d] = acc / l  (fp32)
#pragma unroll
  for (int r = 0; r < 8; ++r) {
    float inv = 1.0f / l[r];
    int row = i * 64 + w * 16 + r + 8 * hl;
#pragma unroll
    for (int n = 0; n < 4; ++n) {
      int col = h * DDIM + n * 16 + ln16;
      out[((size_t)b * TLEN + row) * (HNUM * DDIM) + col] = acc[n][r] * inv;
    }
  }
}

// ---------------------------------------------------------------------------
extern "C" void kernel_launch(void* const* d_in, const int* in_sizes, int n_in,
                              void* d_out, int out_size, void* d_ws,
                              size_t ws_size, hipStream_t stream) {
  const float* x  = (const float*)d_in[0];
  const float* Wq = (const float*)d_in[1];
  const float* Wk = (const float*)d_in[2];
  const float* Wv = (const float*)d_in[3];

  const size_t per = (size_t)BHN * TLEN * DDIM;  // 8M bf16 elems = 16 MB
  __bf16* qws  = (__bf16*)d_ws;
  __bf16* kws  = qws + per;
  __bf16* vtws = kws + per;

  dim3 gProj(TLEN / 64, BHN, 3);
  mha_proj_kernel<<<gProj, 128, 0, stream>>>(x, Wq, Wk, Wv, qws, kws, vtws);

  dim3 gAttn(TLEN / 64, BHN);
  mha_attn_kernel<<<gAttn, 128, 0, stream>>>(qws, kws, vtws, (float*)d_out);
}
